// InferenceModel_11759620457166
// MI455X (gfx1250) — matrix operand order repaired and checked by reading, not compiled, not run on hardware
//
#include <hip/hip_runtime.h>

// ---------------------------------------------------------------------------
// Problem constants (from reference):
//   fea  : (8, 512, 32, 32) f32   -> N = 8192 query rows, D = 512
//   keys : (10000, 512) f32       -> M = 10000
//   out  : (8, 32, 32, 1) f32     -> 8192 values, row n = b*1024 + h*32 + w
// ---------------------------------------------------------------------------
#define DDIM        512
#define N_ROWS      8192
#define M_KEYS      10000
#define M_PAD       10240          // pad to multiple of 32*KC
#define KC          8              // key chunks (grid parallelism over M)
#define M_CHUNK     (M_PAD / KC)   // 1280
#define TILE_M      32             // keys per LDS tile (2 WMMA column tiles)
#define NT          (M_CHUNK / TILE_M) // 40 tiles
#define ROWBLK      128            // q rows per workgroup (8 waves * 16 rows)

// Workspace layout (bytes)
static constexpr size_t OFF_QF   = 0;                                 // N*D*4 f32 q
static constexpr size_t OFF_QB   = OFF_QF  + (size_t)N_ROWS*DDIM*4;   // N*D*2 bf16 q
static constexpr size_t OFF_KB   = OFF_QB  + (size_t)N_ROWS*DDIM*2;   // M_PAD*D*2 bf16 keys
static constexpr size_t OFF_KSQ  = OFF_KB  + (size_t)M_PAD*DDIM*2;    // M_PAD*4
static constexpr size_t OFF_BV   = OFF_KSQ + (size_t)M_PAD*4;         // N*KC*4
static constexpr size_t OFF_BI   = OFF_BV  + (size_t)N_ROWS*KC*4;     // N*KC*4

typedef __attribute__((ext_vector_type(16))) __bf16        v16bf;
typedef __attribute__((ext_vector_type(8)))  float         v8f;
typedef __attribute__((ext_vector_type(4)))  unsigned int  v4u;

static __device__ __forceinline__ unsigned short f32_to_bf16(float x) {
    unsigned int u = __builtin_bit_cast(unsigned int, x);
    u += 0x7FFFu + ((u >> 16) & 1u);       // round to nearest even
    return (unsigned short)(u >> 16);
}

// ---------------------------------------------------------------------------
// Kernel 1: L2-normalize fea over channel dim; emit q as f32 (row-major N x D)
// and bf16 (row-major N x D). One wave per pixel.
// fea element (b,d,h,w) at b*512*1024 + d*1024 + (h*32+w).
// ---------------------------------------------------------------------------
__global__ __launch_bounds__(256) void k_norm(const float* __restrict__ fea,
                                              float* __restrict__ qf,
                                              unsigned short* __restrict__ qb) {
    const int lane = threadIdx.x & 31, wave = threadIdx.x >> 5;
    const int n  = blockIdx.x * 8 + wave;
    const int b  = n >> 10, hw = n & 1023;
    const float* src = fea + (size_t)b * (512 * 1024) + hw;

    float v[16];
    float ss = 0.f;
#pragma unroll
    for (int i = 0; i < 16; ++i) {
        v[i] = src[(size_t)(lane + i * 32) * 1024];
        ss += v[i] * v[i];
    }
#pragma unroll
    for (int m = 16; m >= 1; m >>= 1) ss += __shfl_xor(ss, m, 32);
    const float scale = 1.0f / fmaxf(sqrtf(ss), 1e-12f);

    float*          qrow = qf + (size_t)n * DDIM;
    unsigned short* brow = qb + (size_t)n * DDIM;
#pragma unroll
    for (int i = 0; i < 16; ++i) {
        const float q = v[i] * scale;
        qrow[lane + i * 32] = q;
        brow[lane + i * 32] = f32_to_bf16(q);
    }
}

// ---------------------------------------------------------------------------
// Kernel 2: keys -> bf16 (row-major, padded to M_PAD) + exact f32 ||k||^2.
// Pad rows: zeros, ksq = 1e30 (never selected). One wave per key row.
// ---------------------------------------------------------------------------
__global__ __launch_bounds__(256) void k_keys(const float* __restrict__ keys,
                                              unsigned short* __restrict__ kb,
                                              float* __restrict__ ksq) {
    const int lane = threadIdx.x & 31, wave = threadIdx.x >> 5;
    const int m = blockIdx.x * 8 + wave;
    unsigned short* brow = kb + (size_t)m * DDIM;
    if (m >= M_KEYS) {
#pragma unroll
        for (int i = 0; i < 16; ++i) brow[lane + i * 32] = 0;
        if (lane == 0) ksq[m] = 1.0e30f;
        return;
    }
    const float* src = keys + (size_t)m * DDIM;
    float ss = 0.f;
#pragma unroll
    for (int i = 0; i < 16; ++i) {
        const float v = src[lane + i * 32];
        ss += v * v;
        brow[lane + i * 32] = f32_to_bf16(v);
    }
#pragma unroll
    for (int msk = 16; msk >= 1; msk >>= 1) ss += __shfl_xor(ss, msk, 32);
    if (lane == 0) ksq[m] = ss;
}

// ---------------------------------------------------------------------------
// Kernel 3: fused GEMM (q . k^T via v_wmma_f32_16x16x32_bf16) + running argmin
// of (||k||^2 - 2 q.k). Grid: 64 row-blocks x KC key-chunks. 8 waves/block,
// each wave owns 16 q rows; A tile (16x512 bf16) lives in 128 VGPRs for the
// whole kernel. B tiles of 32 keys (two 16-column WMMA tiles -> two
// INDEPENDENT accumulator chains) double-buffered in LDS, with B fragments
// software-pipelined one K-chunk ahead so v_wmma never waits on dscnt==0.
//
// A frag (16-bit A 16x32, contiguous K as loaded here): lane l holds row M=l%16;
//   lanes 0-15:  VGPR0-3 = K 0..7,  VGPR4-7 = K 8..15
//   lanes 16-31: VGPR0-3 = K 16..23, VGPR4-7 = K 24..31
// B frag (16-bit B 32x16): lane l holds column N=l%16;
//   lanes 0-15 hold K 0..15, lanes 16-31 hold K 16..31 (contiguous K, same order as A)
//   -> a single 32B contiguous LDS load per fragment from key-major storage.
// C/D layout: VGPR r, lane l -> row (r + 8*(l>=16)), col l%16.
// ---------------------------------------------------------------------------
__global__ __launch_bounds__(256) void k_gemm_argmin(
        const unsigned short* __restrict__ qb,
        const unsigned short* __restrict__ kb,
        const float* __restrict__ ksq,
        float* __restrict__ bestv, int* __restrict__ besti) {
    __shared__ unsigned short lds[2][TILE_M * DDIM];  // 2 x 32 KB

    const int lane = threadIdx.x & 31, wave = threadIdx.x >> 5;
    const int rb  = blockIdx.x >> 3;       // row block 0..63
    const int kc  = blockIdx.x & (KC - 1); // key chunk 0..7
    const int row0 = rb * ROWBLK + wave * 16;
    const int sel  = lane >> 4;            // 0: lanes 0-15, 1: lanes 16-31
    const int col  = lane & 15;

    // -------- load register-resident A tile (one global read, reused 40x) ---
    v16bf a[16];
    {
        const unsigned short* arow = qb + (size_t)(row0 + col) * DDIM;
#pragma unroll
        for (int c = 0; c < 16; ++c) {
            union { v4u h[2]; v16bf bf; } u;
            u.h[0] = *(const v4u*)(arow + c * 32 + sel * 16);
            u.h[1] = *(const v4u*)(arow + c * 32 + sel * 16 + 8);
            a[c] = u.bf;
        }
    }

    const int mbase = kc * M_CHUNK;

    // -------- stage tile 0 (32 KB contiguous, 256 threads x 8 x 16B) --------
    {
        const v4u* src = (const v4u*)(kb + (size_t)mbase * DDIM);
        v4u* dst = (v4u*)lds[0];
        for (int i = threadIdx.x; i < 2048; i += 256) dst[i] = src[i];
    }
    __syncthreads();

    float bv[8];
    int   bi[8];
#pragma unroll
    for (int r = 0; r < 8; ++r) { bv[r] = 3.0e38f; bi[r] = 0x7fffffff; }

    for (int t = 0; t < NT; ++t) {
        // prefetch next B tile into registers (overlaps the 32 WMMAs below)
        v4u g[8];
        if (t + 1 < NT) {
            const v4u* src = (const v4u*)(kb + (size_t)(mbase + (t + 1) * TILE_M) * DDIM);
#pragma unroll
            for (int i = 0; i < 8; ++i) g[i] = src[threadIdx.x + i * 256];
        }

        // two independent accumulator chains over K = 512 (16 chunks of 32),
        // B fragments pipelined one chunk ahead
        const unsigned short* bcol0 = lds[t & 1] + col * DDIM + sel * 16;
        const unsigned short* bcol1 = bcol0 + 16 * DDIM;
        v8f acc0 = {}, acc1 = {};
        v16bf b0 = *(const v16bf*)(bcol0);
        v16bf b1 = *(const v16bf*)(bcol1);
#pragma unroll
        for (int c = 0; c < 16; ++c) {
            v16bf nb0, nb1;
            if (c + 1 < 16) {
                nb0 = *(const v16bf*)(bcol0 + (c + 1) * 32);
                nb1 = *(const v16bf*)(bcol1 + (c + 1) * 32);
            }
            acc0 = __builtin_amdgcn_wmma_f32_16x16x32_bf16(
                       false, a[c], false, b0, (short)0, acc0, false, false);
            acc1 = __builtin_amdgcn_wmma_f32_16x16x32_bf16(
                       false, a[c], false, b1, (short)0, acc1, false, false);
            b0 = nb0; b1 = nb1;
        }

        // running argmin of (||k||^2 - 2 q.k); q_sq==1 is constant, ignored
        const int   mcol0 = mbase + t * TILE_M + col;
        const int   mcol1 = mcol0 + 16;
        const float kq0   = ksq[mcol0];
        const float kq1   = ksq[mcol1];
#pragma unroll
        for (int r = 0; r < 8; ++r) {
            const float d0 = kq0 - 2.0f * acc0[r];
            if (d0 < bv[r]) { bv[r] = d0; bi[r] = mcol0; }
            const float d1 = kq1 - 2.0f * acc1[r];
            if (d1 < bv[r]) { bv[r] = d1; bi[r] = mcol1; }
        }

        if (t + 1 < NT) {
            __syncthreads();               // everyone done reading old buffer
            v4u* dst = (v4u*)lds[(t + 1) & 1];
#pragma unroll
            for (int i = 0; i < 8; ++i) dst[threadIdx.x + i * 256] = g[i];
            __syncthreads();
        }
    }

    // -------- butterfly argmin across the 16 lanes sharing a row set --------
#pragma unroll
    for (int r = 0; r < 8; ++r) {
        float v = bv[r]; int ix = bi[r];
#pragma unroll
        for (int m = 1; m < 16; m <<= 1) {
            const float ov = __shfl_xor(v, m, 32);
            const int   oi = __shfl_xor(ix, m, 32);
            if (ov < v || (ov == v && oi < ix)) { v = ov; ix = oi; }
        }
        bv[r] = v; bi[r] = ix;
    }
    if (col == 0) {                         // lanes 0 (rows 0-7) and 16 (rows 8-15)
        const int rowbase = row0 + sel * 8;
#pragma unroll
        for (int r = 0; r < 8; ++r) {
            bestv[(size_t)(rowbase + r) * KC + kc] = bv[r];
            besti[(size_t)(rowbase + r) * KC + kc] = bi[r];
        }
    }
}

// ---------------------------------------------------------------------------
// Kernel 4: reduce KC partial argmins, gather winning f32 key, exact f32
// heatmap = sum_d (q - k)^4. One wave per output row.
// ---------------------------------------------------------------------------
__global__ __launch_bounds__(256) void k_heat(const float* __restrict__ qf,
                                              const float* __restrict__ keys,
                                              const float* __restrict__ bestv,
                                              const int* __restrict__ besti,
                                              float* __restrict__ out) {
    const int lane = threadIdx.x & 31, wave = threadIdx.x >> 5;
    const int n = blockIdx.x * 8 + wave;

    float v = bestv[(size_t)n * KC];
    int   ix = besti[(size_t)n * KC];
#pragma unroll
    for (int c = 1; c < KC; ++c) {
        const float ov = bestv[(size_t)n * KC + c];
        const int   oi = besti[(size_t)n * KC + c];
        if (ov < v || (ov == v && oi < ix)) { v = ov; ix = oi; }
    }

    const float* q = qf   + (size_t)n * DDIM;
    const float* k = keys + (size_t)ix * DDIM;
    float s = 0.f;
#pragma unroll
    for (int i = 0; i < 16; ++i) {
        const float d  = q[lane + i * 32] - k[lane + i * 32];
        const float d2 = d * d;
        s += d2 * d2;
    }
#pragma unroll
    for (int m = 16; m >= 1; m >>= 1) s += __shfl_xor(s, m, 32);
    if (lane == 0) out[n] = s;
}

// ---------------------------------------------------------------------------
extern "C" void kernel_launch(void* const* d_in, const int* in_sizes, int n_in,
                              void* d_out, int out_size, void* d_ws, size_t ws_size,
                              hipStream_t stream) {
    const float* fea  = (const float*)d_in[0];
    const float* keys = (const float*)d_in[1];
    char* ws = (char*)d_ws;

    float*          qf  = (float*)(ws + OFF_QF);
    unsigned short* qb  = (unsigned short*)(ws + OFF_QB);
    unsigned short* kb  = (unsigned short*)(ws + OFF_KB);
    float*          ksq = (float*)(ws + OFF_KSQ);
    float*          bv  = (float*)(ws + OFF_BV);
    int*            bi  = (int*)(ws + OFF_BI);
    float*          out = (float*)d_out;

    k_norm<<<N_ROWS / 8, 256, 0, stream>>>(fea, qf, qb);
    k_keys<<<M_PAD / 8, 256, 0, stream>>>(keys, kb, ksq);
    k_gemm_argmin<<<(N_ROWS / ROWBLK) * KC, 256, 0, stream>>>(qb, kb, ksq, bv, bi);
    k_heat<<<N_ROWS / 8, 256, 0, stream>>>(qf, keys, bv, bi, out);
}
